// MDGAT_41094247088150
// MI455X (gfx1250) — compile-verified, matrix operand order checked by reading, not run on hardware
//
#include <hip/hip_runtime.h>
#include <hip/hip_bf16.h>

typedef __attribute__((ext_vector_type(16))) _Float16 v16h;
typedef __attribute__((ext_vector_type(8)))  _Float16 v8h;
typedef __attribute__((ext_vector_type(4)))  _Float16 v4h;
typedef __attribute__((ext_vector_type(8)))  float    v8f;
typedef __attribute__((ext_vector_type(4)))  float    v4f;

#define NTOK 1024
#define DMODEL 256
#define HEADS 4
#define DHEAD 64
#define NL 18
#define SLICES 4   // 2 sides x 2 batch
#define MP1 1025   // 1024+1 (dustbin)

// -------- CDNA5 WMMA fragment loader (16-bit A/B, 16x32 / 32x16) ----------
// LDS tile is [rows][ldk] halves, K contiguous; koff selects the 32-K slab.
// Per ISA 7.12.2: lane m = lane&15, octet h = lane>>4;
// halves 0..7 -> K = koff + 8h + e, halves 8..15 -> K = koff + 16 + 8h + e.
__device__ __forceinline__ v16h frag_ld(const _Float16* tile, int row, int ldk,
                                        int koff) {
  const int lane = threadIdx.x & 31;
  const int h = lane >> 4;
  const _Float16* p = tile + (row + (lane & 15)) * ldk + koff;
  v16h f;
#pragma unroll
  for (int e = 0; e < 8; ++e) f[e] = p[8 * h + e];
#pragma unroll
  for (int e = 0; e < 8; ++e) f[8 + e] = p[16 + 8 * h + e];
  return f;
}

// ---------------- block reductions (wave32) ----------------
__device__ __forceinline__ float blk_max(float v, float* sh) {
#pragma unroll
  for (int o = 16; o > 0; o >>= 1) v = fmaxf(v, __shfl_xor(v, o, 32));
  int w = threadIdx.x >> 5, lane = threadIdx.x & 31;
  if (lane == 0) sh[w] = v;
  __syncthreads();
  int nw = blockDim.x >> 5;
  v = (lane < nw) ? sh[lane] : -3.4e38f;
  if (w == 0) {
#pragma unroll
    for (int o = 16; o > 0; o >>= 1) v = fmaxf(v, __shfl_xor(v, o, 32));
    if (lane == 0) sh[0] = v;
  }
  __syncthreads();
  v = sh[0];
  __syncthreads();
  return v;
}

__device__ __forceinline__ float blk_sum(float v, float* sh) {
#pragma unroll
  for (int o = 16; o > 0; o >>= 1) v += __shfl_xor(v, o, 32);
  int w = threadIdx.x >> 5, lane = threadIdx.x & 31;
  if (lane == 0) sh[w] = v;
  __syncthreads();
  int nw = blockDim.x >> 5;
  v = (lane < nw) ? sh[lane] : 0.f;
  if (w == 0) {
#pragma unroll
    for (int o = 16; o > 0; o >>= 1) v += __shfl_xor(v, o, 32);
    if (lane == 0) sh[0] = v;
  }
  __syncthreads();
  v = sh[0];
  __syncthreads();
  return v;
}

// ---------------- batched GEMM: Y = epilogue(W @ X + b) -------------------
// W [cout, cin] f32, X [slice][cin][1024] f32, Y [slice][out_cs][1024] f32.
// Block: 256 thr (8 waves), tile 64(co) x 64(j), K staged 64 wide.
// Wave w: m_sub=(w&1)*32 (two 16-row tiles), n_sub=(w>>1)*16.
// 4 WMMAs per wave per staging round. Load-clause then convert/store staging.
__global__ __launch_bounds__(256) void gemm_kernel(
    const float* __restrict__ W, const float* __restrict__ bias,
    const float* __restrict__ X, float* __restrict__ Y,
    const float* __restrict__ res, const float* __restrict__ bnG,
    const float* __restrict__ bnB,
    int cin, int mode, int src_xor, int out_cs, int out_coff) {
  __shared__ _Float16 Wt[64 * 64];
  __shared__ _Float16 Xt[64 * 64];
  const int j0 = blockIdx.x * 64;
  const int co0 = blockIdx.y * 64;
  const int sl = blockIdx.z;
  const float* Xs = X + (size_t)(sl ^ src_xor) * cin * NTOK;
  const int t = threadIdx.x, w = t >> 5;
  const int m_sub = (w & 1) * 32, n_sub = (w >> 1) * 16;
  v8f acc0 = {}, acc1 = {};
  for (int k0 = 0; k0 < cin; k0 += 64) {
    // Phase 1: issue all global b128 loads (one clause, latency overlapped)
    v4f wv[4], xv[4];
#pragma unroll
    for (int e = 0; e < 4; ++e) {
      int g = t + 256 * e, r = g >> 4, c4 = (g & 15) * 4;
      wv[e] = *(const v4f*)&W[(size_t)(co0 + r) * cin + k0 + c4];
    }
#pragma unroll
    for (int e = 0; e < 4; ++e) {
      int g = t + 256 * e, jj4 = (g & 15) * 4, c = g >> 4;
      xv[e] = *(const v4f*)&Xs[(size_t)(k0 + c) * NTOK + j0 + jj4];
    }
    // Phase 2: convert + LDS stores
#pragma unroll
    for (int e = 0; e < 4; ++e) {
      int g = t + 256 * e, r = g >> 4, c4 = (g & 15) * 4;
      *(v4h*)&Wt[r * 64 + c4] = __builtin_convertvector(wv[e], v4h);
    }
#pragma unroll
    for (int e = 0; e < 4; ++e) {
      int g = t + 256 * e, jj4 = (g & 15) * 4, c = g >> 4;
#pragma unroll
      for (int q = 0; q < 4; ++q) Xt[(jj4 + q) * 64 + c] = (_Float16)xv[e][q];
    }
    if (k0 + 64 < cin) {  // prefetch next K slab (global_prefetch_b8)
      __builtin_prefetch(&W[(size_t)(co0 + (t >> 6)) * cin + k0 + 64 + (t & 63)], 0, 0);
      __builtin_prefetch(&Xs[(size_t)(k0 + 64 + (t >> 6)) * NTOK + j0 + (t & 63)], 0, 0);
    }
    __syncthreads();
#pragma unroll
    for (int kk = 0; kk < 64; kk += 32) {
      v16h b = frag_ld(Xt, n_sub, 64, kk);
      v16h a0 = frag_ld(Wt, m_sub, 64, kk);
      v16h a1 = frag_ld(Wt, m_sub + 16, 64, kk);
      acc0 = __builtin_amdgcn_wmma_f32_16x16x32_f16(false, a0, false, b,
                                                    (short)0, acc0, false, false);
      acc1 = __builtin_amdgcn_wmma_f32_16x16x32_f16(false, a1, false, b,
                                                    (short)0, acc1, false, false);
    }
    __syncthreads();
  }
  const int lane = t & 31;
  const int jl = n_sub + (lane & 15);
  const int hh = lane >> 4;
#pragma unroll
  for (int part = 0; part < 2; ++part) {
#pragma unroll
    for (int r = 0; r < 8; ++r) {
      int co = co0 + m_sub + part * 16 + r + 8 * hh;
      float v = (part ? acc1[r] : acc0[r]) + bias[co];
      if (mode == 1)
        v = fmaxf(0.f, v * (bnG[co] * rsqrtf(1.f + 1e-5f)) + bnB[co]);
      size_t oi = (size_t)sl * out_cs * NTOK + (size_t)(out_coff + co) * NTOK +
                  (j0 + jl);
      if (mode == 2) v += res[oi];
      Y[oi] = v;
    }
  }
}

// ------------- scores: S[z,i,j] = scale * sum_d Q[ch(d),i] * K[ch(d),j] ----
// ch(d) = d*hstride + (z % H). Attention: D=64,hstride=4,H=4 (one staging
// round). Final matcher: D=256,hstride=1,H=1. Block: 16(i) x 128(j), 8 waves,
// K staged 64 wide -> 2 WMMAs per wave per round.
__global__ __launch_bounds__(256) void scores_kernel(
    const float* __restrict__ Q, const float* __restrict__ Kb,
    float* __restrict__ S, int D, int hstride, int H, float scale) {
  __shared__ _Float16 At[16 * 64];
  __shared__ _Float16 Bt[128 * 64];
  const int j0 = blockIdx.x * 128;
  const int i0 = blockIdx.y * 16;
  const int z = blockIdx.z;
  const int sl = z / H, h = z % H;
  const float* Qs = Q + (size_t)sl * (D * hstride) * NTOK;
  const float* Ks = Kb + (size_t)sl * (D * hstride) * NTOK;
  const int t = threadIdx.x, w = t >> 5;
  const int j_sub = w * 16;
  v8f acc = {};
  for (int d0 = 0; d0 < D; d0 += 64) {
    // Phase 1: global b128 load clause
    v4f qv, kv[8];
    {
      int m4 = (t & 3) * 4, c = t >> 2;
      qv = *(const v4f*)&Qs[(size_t)((d0 + c) * hstride + h) * NTOK + i0 + m4];
    }
#pragma unroll
    for (int e = 0; e < 8; ++e) {
      int g = t + 256 * e, jj4 = (g & 31) * 4, c = g >> 5;
      kv[e] = *(const v4f*)&Ks[(size_t)((d0 + c) * hstride + h) * NTOK + j0 + jj4];
    }
    // Phase 2: convert + transpose-store
    {
      int m4 = (t & 3) * 4, c = t >> 2;
#pragma unroll
      for (int q = 0; q < 4; ++q) At[(m4 + q) * 64 + c] = (_Float16)qv[q];
    }
#pragma unroll
    for (int e = 0; e < 8; ++e) {
      int g = t + 256 * e, jj4 = (g & 31) * 4, c = g >> 5;
#pragma unroll
      for (int q = 0; q < 4; ++q) Bt[(jj4 + q) * 64 + c] = (_Float16)kv[e][q];
    }
    __syncthreads();
#pragma unroll
    for (int kk = 0; kk < 64; kk += 32) {
      v16h a = frag_ld(At, 0, 64, kk);
      v16h b = frag_ld(Bt, j_sub, 64, kk);
      acc = __builtin_amdgcn_wmma_f32_16x16x32_f16(false, a, false, b, (short)0,
                                                   acc, false, false);
    }
    __syncthreads();
  }
  const int lane = t & 31;
  const int jl = j_sub + (lane & 15);
  const int hh = lane >> 4;
#pragma unroll
  for (int r = 0; r < 8; ++r) {
    int ii = i0 + r + 8 * hh;
    S[((size_t)z * NTOK + ii) * NTOK + j0 + jl] = acc[r] * scale;
  }
}

// ------------- full softmax over 1024 keys; writes f16 probs --------------
__global__ __launch_bounds__(256) void softmax_kernel(
    const float* __restrict__ S, _Float16* __restrict__ P) {
  __shared__ float sh[8];
  const size_t row = blockIdx.x;
  const float* s = S + row * NTOK;
  _Float16* p = P + row * NTOK;
  const int t = threadIdx.x;
  float v[4];
#pragma unroll
  for (int e = 0; e < 4; ++e) v[e] = s[t + 256 * e];
  float mx = fmaxf(fmaxf(v[0], v[1]), fmaxf(v[2], v[3]));
  mx = blk_max(mx, sh);
  float sum = 0.f;
#pragma unroll
  for (int e = 0; e < 4; ++e) { v[e] = __expf(v[e] - mx); sum += v[e]; }
  sum = blk_sum(sum, sh);
  float inv = 1.f / sum;
#pragma unroll
  for (int e = 0; e < 4; ++e) p[t + 256 * e] = (_Float16)(v[e] * inv);
}

// ------------- top-k softmax: exact radix select on ordered float bits ----
__device__ __forceinline__ unsigned fkey(float f) {
  unsigned u = __float_as_uint(f);
  return (u & 0x80000000u) ? ~u : (u | 0x80000000u);
}

__global__ __launch_bounds__(256) void softmax_topk_kernel(
    const float* __restrict__ S, _Float16* __restrict__ P, int K) {
  __shared__ unsigned hist[256];
  __shared__ unsigned sh_prefix;
  __shared__ int sh_rem, tie_ctr;
  __shared__ float sh[8];
  const size_t row = blockIdx.x;
  const float* s = S + row * NTOK;
  _Float16* p = P + row * NTOK;
  const int t = threadIdx.x;
  float v[4];
  unsigned key[4];
#pragma unroll
  for (int e = 0; e < 4; ++e) { v[e] = s[t + 256 * e]; key[e] = fkey(v[e]); }
  unsigned prefix = 0;
  int rem = K;
  for (int shift = 24; shift >= 0; shift -= 8) {
    hist[t] = 0;
    __syncthreads();
#pragma unroll
    for (int e = 0; e < 4; ++e) {
      unsigned kk = key[e];
      bool act = (shift == 24) || ((kk >> (shift + 8)) == prefix);
      if (act) atomicAdd(&hist[(kk >> shift) & 255], 1u);
    }
    __syncthreads();
    if (t == 0) {
      int cum = 0, b;
      for (b = 255; b >= 0; --b) {
        cum += (int)hist[b];
        if (cum >= rem) break;
      }
      if (b < 0) b = 0;
      sh_rem = rem - (cum - (int)hist[b]);
      sh_prefix = (prefix << 8) | (unsigned)b;
    }
    __syncthreads();
    prefix = sh_prefix;
    rem = sh_rem;
    __syncthreads();
  }
  if (t == 0) tie_ctr = rem;
  __syncthreads();
  bool sel[4];
  float mx = -3.4e38f;
#pragma unroll
  for (int e = 0; e < 4; ++e) {
    if (key[e] > prefix) sel[e] = true;
    else if (key[e] == prefix) sel[e] = (atomicSub(&tie_ctr, 1) > 0);
    else sel[e] = false;
    if (sel[e]) mx = fmaxf(mx, v[e]);
  }
  mx = blk_max(mx, sh);
  float wv[4];
  float sum = 0.f;
#pragma unroll
  for (int e = 0; e < 4; ++e) {
    wv[e] = sel[e] ? __expf(v[e] - mx) : 0.f;
    sum += wv[e];
  }
  sum = blk_sum(sum, sh);
  float inv = 1.f / sum;
#pragma unroll
  for (int e = 0; e < 4; ++e) p[t + 256 * e] = (_Float16)(wv[e] * inv);
}

// ------------- PV: O[ch(d),i] = sum_j P[i,j] * V[ch(d),j] -----------------
// Block: 128 thr (4 waves), tile 16(i) x 64(d), K staged 64 wide over j.
// 2 WMMAs per wave per staging round. Fully vectorized staging (P is f16).
__global__ __launch_bounds__(128) void pv_kernel(
    const _Float16* __restrict__ P, const float* __restrict__ V,
    float* __restrict__ O) {
  __shared__ _Float16 At[16 * 64];
  __shared__ _Float16 Bt[64 * 64];
  const int i0 = blockIdx.x * 16;
  const int z = blockIdx.y;  // sl*4 + h
  const int sl = z >> 2, h = z & 3;
  const _Float16* Pr = P + ((size_t)z * NTOK + i0) * NTOK;
  const float* Vs = V + (size_t)sl * DMODEL * NTOK;
  const int t = threadIdx.x, w = t >> 5;
  const int d_sub = w * 16;
  v8f acc = {};
  for (int j0 = 0; j0 < NTOK; j0 += 64) {
    // Phase 1: global load clause
    v8h pv8;
    v4f vv[8];
    {
      int c8 = (t & 7) * 8, m = t >> 3;
      pv8 = *(const v8h*)&Pr[(size_t)m * NTOK + j0 + c8];
    }
#pragma unroll
    for (int e = 0; e < 8; ++e) {
      int g = t + 128 * e, c4 = (g & 15) * 4, d = g >> 4;
      vv[e] = *(const v4f*)&Vs[(size_t)(d * 4 + h) * NTOK + j0 + c4];
    }
    // Phase 2: convert + LDS stores
    {
      int c8 = (t & 7) * 8, m = t >> 3;
      *(v8h*)&At[m * 64 + c8] = pv8;
    }
#pragma unroll
    for (int e = 0; e < 8; ++e) {
      int g = t + 128 * e, c4 = (g & 15) * 4, d = g >> 4;
      *(v4h*)&Bt[d * 64 + c4] = __builtin_convertvector(vv[e], v4h);
    }
    if (j0 + 64 < NTOK) {
      __builtin_prefetch(&Vs[(size_t)((t >> 6) * 4 + h) * NTOK + j0 + 64 + (t & 63)], 0, 0);
    }
    __syncthreads();
#pragma unroll
    for (int kk = 0; kk < 64; kk += 32) {
      v16h a = frag_ld(At, 0, 64, kk);
      v16h b = frag_ld(Bt, d_sub, 64, kk);
      acc = __builtin_amdgcn_wmma_f32_16x16x32_f16(false, a, false, b, (short)0,
                                                   acc, false, false);
    }
    __syncthreads();
  }
  const int lane = t & 31;
  const int dl = d_sub + (lane & 15);
  const int hh = lane >> 4;
#pragma unroll
  for (int r = 0; r < 8; ++r) {
    int ii = i0 + r + 8 * hh;
    O[(size_t)sl * DMODEL * NTOK + (size_t)(dl * 4 + h) * NTOK + ii] = acc[r];
  }
}

// ------------- misc small kernels -----------------------------------------
__global__ void copy_x_to_h(const float* __restrict__ x, float* __restrict__ h) {
  size_t idx = (size_t)blockIdx.x * 256 + threadIdx.x;
  if (idx < (size_t)SLICES * DMODEL * NTOK) {
    size_t sl = idx >> 18;                 // 256*1024 = 2^18
    size_t rm = idx & ((1u << 18) - 1);
    h[sl * (512 * NTOK) + rm] = x[idx];
  }
}

__global__ void couplings_init(const float* __restrict__ S,
                               const float* __restrict__ alpha,
                               float* __restrict__ C, float* __restrict__ u,
                               float* __restrict__ v) {
  size_t idx = (size_t)blockIdx.x * 256 + threadIdx.x;
  const size_t total = (size_t)2 * MP1 * MP1;
  if (idx < total) {
    int b = (int)(idx / ((size_t)MP1 * MP1));
    int rm = (int)(idx % ((size_t)MP1 * MP1));
    int i = rm / MP1, j = rm % MP1;
    C[idx] = (i < NTOK && j < NTOK)
                 ? S[((size_t)b * NTOK + i) * NTOK + j]
                 : alpha[0];
  }
  if (idx < (size_t)2 * MP1) { u[idx] = 0.f; v[idx] = 0.f; }
}

__global__ __launch_bounds__(256) void sink_u(const float* __restrict__ C,
                                              const float* __restrict__ v,
                                              float* __restrict__ u) {
  __shared__ float sh[8];
  const int i = blockIdx.x, b = blockIdx.y;
  const float* row = C + ((size_t)b * MP1 + i) * MP1;
  const float* vb = v + (size_t)b * MP1;
  const int t = threadIdx.x;
  float mx = -3.4e38f;
  for (int j = t; j < MP1; j += 256) mx = fmaxf(mx, row[j] + vb[j]);
  mx = blk_max(mx, sh);
  float sum = 0.f;
  for (int j = t; j < MP1; j += 256) sum += __expf(row[j] + vb[j] - mx);
  sum = blk_sum(sum, sh);
  if (t == 0) {
    const float norm = -logf(2048.f);
    float lmu = (i < NTOK) ? norm : (logf(1024.f) + norm);
    u[(size_t)b * MP1 + i] = lmu - (mx + logf(sum));
  }
}

__global__ __launch_bounds__(256) void sink_v(const float* __restrict__ C,
                                              const float* __restrict__ u,
                                              float* __restrict__ v) {
  __shared__ float sh[8];
  const int j = blockIdx.x, b = blockIdx.y;
  const float* col = C + (size_t)b * MP1 * MP1 + j;
  const float* ub = u + (size_t)b * MP1;
  const int t = threadIdx.x;
  float mx = -3.4e38f;
  for (int i = t; i < MP1; i += 256) mx = fmaxf(mx, col[(size_t)i * MP1] + ub[i]);
  mx = blk_max(mx, sh);
  float sum = 0.f;
  for (int i = t; i < MP1; i += 256) sum += __expf(col[(size_t)i * MP1] + ub[i] - mx);
  sum = blk_sum(sum, sh);
  if (t == 0) {
    const float norm = -logf(2048.f);
    float lnu = (j < NTOK) ? norm : (logf(1024.f) + norm);
    v[(size_t)b * MP1 + j] = lnu - (mx + logf(sum));
  }
}

__global__ void final_kernel(const float* __restrict__ C,
                             const float* __restrict__ u,
                             const float* __restrict__ v,
                             float* __restrict__ out) {
  size_t idx = (size_t)blockIdx.x * 256 + threadIdx.x;
  const size_t total = (size_t)2 * MP1 * MP1;
  if (idx < total) {
    int b = (int)(idx / ((size_t)MP1 * MP1));
    int rm = (int)(idx % ((size_t)MP1 * MP1));
    int i = rm / MP1, j = rm % MP1;
    out[idx] = C[idx] + u[(size_t)b * MP1 + i] + v[(size_t)b * MP1 + j] +
               logf(2048.f);
  }
}

// --------------------------------------------------------------------------
extern "C" void kernel_launch(void* const* d_in, const int* in_sizes, int n_in,
                              void* d_out, int out_size, void* d_ws,
                              size_t ws_size, hipStream_t stream) {
  const float* desc0   = (const float*)d_in[0];
  const float* desc1   = (const float*)d_in[1];
  const float* proj_w  = (const float*)d_in[2];
  const float* proj_b  = (const float*)d_in[3];
  const float* merge_w = (const float*)d_in[4];
  const float* merge_b = (const float*)d_in[5];
  const float* mlp1_w  = (const float*)d_in[6];
  const float* mlp1_b  = (const float*)d_in[7];
  const float* bn_g    = (const float*)d_in[8];
  const float* bn_b    = (const float*)d_in[9];
  const float* mlp2_w  = (const float*)d_in[10];
  const float* mlp2_b  = (const float*)d_in[11];
  const float* final_w = (const float*)d_in[12];
  const float* final_b = (const float*)d_in[13];
  const float* alpha   = (const float*)d_in[14];
  float* out = (float*)d_out;

  // ---- workspace carve (floats) ----
  float* ws = (float*)d_ws;
  const size_t SLICE = (size_t)DMODEL * NTOK;  // 262144
  float* x    = ws;               // [4][256][1024]
  float* q    = x    + 4 * SLICE;
  float* kbuf = q    + 4 * SLICE;
  float* vbuf = kbuf + 4 * SLICE;
  float* obuf = vbuf + 4 * SLICE; // attn out / reused as final-proj out
  float* hbuf = obuf + 4 * SLICE;           // [4][512][1024]
  float* tbuf = hbuf + 8 * SLICE;           // [4][512][1024]
  float* Sbuf = tbuf + 8 * SLICE;           // [16][1024][1024] scores
  _Float16* Pbuf = (_Float16*)(Sbuf + (size_t)16 * NTOK * NTOK);  // f16 probs
  // Sinkhorn buffers reuse the Sbuf region after the GNN layers:
  float* mS = Sbuf;                                   // [2][1024][1024]
  float* C  = Sbuf + (size_t)2 * NTOK * NTOK;         // [2][1025][1025]
  float* uu = C + (size_t)2 * MP1 * MP1;              // [2][1025]
  float* vv = uu + 2 * MP1;                           // [2][1025]

  // x[0:2] = desc0, x[2:4] = desc1 (side-major slices)
  hipMemcpyAsync(x, desc0, 2 * SLICE * sizeof(float), hipMemcpyDeviceToDevice,
                 stream);
  hipMemcpyAsync(x + 2 * SLICE, desc1, 2 * SLICE * sizeof(float),
                 hipMemcpyDeviceToDevice, stream);

  static const int KLIST[6] = {128, 128, 64, 64, 32, 32};
  const dim3 g256(16, 4, SLICES);   // cout=256 GEMMs (64x64 tiles)
  const dim3 g512(16, 8, SLICES);   // cout=512 GEMM
  const dim3 gS(8, 64, 16);         // attention scores
  const dim3 gPV(64, 16);

  for (int i = 0; i < NL; ++i) {
    const bool cross = (i & 1) != 0;
    const int sx = cross ? 2 : 0;               // side-swap for k/v source
    const int topk = (i > NL - 1 - 6) ? KLIST[i - (NL - 6)] : -1;
    const float* pw0 = proj_w + ((size_t)i * 3 + 0) * 256 * 256;
    const float* pw1 = proj_w + ((size_t)i * 3 + 1) * 256 * 256;
    const float* pw2 = proj_w + ((size_t)i * 3 + 2) * 256 * 256;
    const float* pb0 = proj_b + ((size_t)i * 3 + 0) * 256;
    const float* pb1 = proj_b + ((size_t)i * 3 + 1) * 256;
    const float* pb2 = proj_b + ((size_t)i * 3 + 2) * 256;
    const float* mw = merge_w + (size_t)i * 256 * 256;
    const float* mb = merge_b + (size_t)i * 256;
    const float* w1 = mlp1_w + (size_t)i * 512 * 512;
    const float* b1 = mlp1_b + (size_t)i * 512;
    const float* g1 = bn_g + (size_t)i * 512;
    const float* be1 = bn_b + (size_t)i * 512;
    const float* w2 = mlp2_w + (size_t)i * 256 * 512;
    const float* b2 = mlp2_b + (size_t)i * 256;

    gemm_kernel<<<g256, 256, 0, stream>>>(pw0, pb0, x, q, nullptr, nullptr,
                                          nullptr, 256, 0, 0, 256, 0);
    gemm_kernel<<<g256, 256, 0, stream>>>(pw1, pb1, x, kbuf, nullptr, nullptr,
                                          nullptr, 256, 0, sx, 256, 0);
    gemm_kernel<<<g256, 256, 0, stream>>>(pw2, pb2, x, vbuf, nullptr, nullptr,
                                          nullptr, 256, 0, sx, 256, 0);
    scores_kernel<<<gS, 256, 0, stream>>>(q, kbuf, Sbuf, DHEAD, HEADS, HEADS,
                                          0.125f);
    if (topk < 0)
      softmax_kernel<<<16 * NTOK, 256, 0, stream>>>(Sbuf, Pbuf);
    else
      softmax_topk_kernel<<<16 * NTOK, 256, 0, stream>>>(Sbuf, Pbuf, topk);
    pv_kernel<<<gPV, 128, 0, stream>>>(Pbuf, vbuf, obuf);
    // msg -> upper half of h; x -> lower half of h
    gemm_kernel<<<g256, 256, 0, stream>>>(mw, mb, obuf, hbuf, nullptr, nullptr,
                                          nullptr, 256, 0, 0, 512, 256);
    copy_x_to_h<<<(int)((4 * SLICE + 255) / 256), 256, 0, stream>>>(x, hbuf);
    gemm_kernel<<<g512, 256, 0, stream>>>(w1, b1, hbuf, tbuf, nullptr, g1, be1,
                                          512, 1, 0, 512, 0);
    gemm_kernel<<<g256, 256, 0, stream>>>(w2, b2, tbuf, x, x, nullptr, nullptr,
                                          512, 2, 0, 256, 0);
  }

  // final projection (all 4 slices), matcher scores, Sinkhorn
  gemm_kernel<<<g256, 256, 0, stream>>>(final_w, final_b, x, obuf, nullptr,
                                        nullptr, nullptr, 256, 0, 0, 256, 0);
  scores_kernel<<<dim3(8, 64, 2), 256, 0, stream>>>(
      obuf, obuf + 2 * SLICE, mS, DMODEL, 1, 1, 1.f / 16.f);

  const int nC = (int)(((size_t)2 * MP1 * MP1 + 255) / 256);
  couplings_init<<<nC, 256, 0, stream>>>(mS, alpha, C, uu, vv);
  for (int it = 0; it < 100; ++it) {
    sink_u<<<dim3(MP1, 2), 256, 0, stream>>>(C, vv, uu);
    sink_v<<<dim3(MP1, 2), 256, 0, stream>>>(C, uu, vv);
  }
  final_kernel<<<nC, 256, 0, stream>>>(C, uu, vv, out);
}
